// SimpleEncoder_87995289960501
// MI455X (gfx1250) — matrix operand order repaired and checked by reading, not compiled, not run on hardware
//
#include <hip/hip_runtime.h>
#include <hip/hip_bf16.h>

// ---------------------------------------------------------------------------
// SimpleEncoder (LSTM-512 over T=512, B=512) for MI455X / gfx1250 (wave32).
//
//   * persistent kernel, 64 blocks (8x8 tiling of [B=512] x [N=512])
//   * recurrent weights [w_h; w_x] bf16 WMMA fragments, 288KB slice staged
//     into LDS via the Tensor Data Mover (tensor_load_to_lds + tensorcnt)
//   * c-state in VGPRs for all 512 steps; h ping-pongs through L2
//   * v_wmma_f32_16x16x32_bf16, fp32 accumulate, fused gate math
//   * AS(1) casts -> global_load_b128 (LOADcnt only, no DScnt coupling)
//   * native __bf16 converts (v_cvt_*_bf16) instead of VALU bit-emulation
//   * global_prefetch of next step's x_t rows; grid barrier with s_sleep
// ---------------------------------------------------------------------------

typedef __attribute__((ext_vector_type(16))) __bf16 v16bf;
typedef __attribute__((ext_vector_type(8)))  float  v8f;
typedef unsigned int u32x4 __attribute__((ext_vector_type(4)));
typedef int          i32x8 __attribute__((ext_vector_type(8)));
typedef int          i32x4 __attribute__((ext_vector_type(4)));

// primitive 128-bit vectors (trivial types -> legal through AS(1) pointers)
typedef unsigned int u4v __attribute__((ext_vector_type(4)));
typedef float        f4v __attribute__((ext_vector_type(4)));

// global (AS1) pointee typedefs
typedef __attribute__((address_space(1))) const u4v            g_cu4v;
typedef __attribute__((address_space(1))) const f4v            g_cf4v;
typedef __attribute__((address_space(1))) const float          g_cfloat;
typedef __attribute__((address_space(1))) const unsigned short g_cushort;
typedef __attribute__((address_space(1))) unsigned short       g_ushort;
typedef __attribute__((address_space(1))) float                g_float;
typedef __attribute__((address_space(1))) unsigned             g_uint;

#define BATCH   512
#define NH      512     // hidden
#define NF      64      // input features / step
#define TSTEPS  512
#define NKT     18      // K tiles of 32 (16 from h, 2 from x_t)
#define NZT     128     // z-col tiles of 16 (4*NH/16)
#define NBLK    64      // persistent blocks (8 x 8)

union FragU {
    v16bf v;
    u4v   q[2];
    unsigned u[8];
};

// native hardware f32->bf16 (RNE) -- lets clang emit v_cvt_*_bf16
__device__ __forceinline__ unsigned short bf16bits(float f) {
    __bf16 h = (__bf16)f;
    return __builtin_bit_cast(unsigned short, h);
}
__device__ __forceinline__ unsigned pk2(float lo, float hi) {
    return (unsigned)bf16bits(lo) | ((unsigned)bf16bits(hi) << 16);
}
__device__ __forceinline__ float sigmoidf_(float x) {
    return 1.0f / (1.0f + __expf(-x));
}
__device__ __forceinline__ float seluf_(float x) {
    const float sc = 1.0507009873554805f, al = 1.6732632423543772f;
    return x > 0.0f ? sc * x : sc * al * (__expf(x) - 1.0f);
}

// -------- prep: pack [w_h ; w_x] (576 x 2048 f32) into bf16 WMMA-B frags ----
// Fragment (kt, nt): 32x16 tile of W, 1024B. Per-lane layout (matches reads):
//   lane l, half h=l>>4: column n = nt*16 + (l&15),
//   K = kt*32 + h*8 + e (e<8), kt*32 + 16 + h*8 + (e-8) (e>=8)
//   bytes: e<8 -> frag + l*16 + 2e ; e>=8 -> frag + 512 + l*16 + 2(e-8)
__global__ void pack_w_kernel(const float* __restrict__ w_x,
                              const float* __restrict__ w_h,
                              unsigned short* __restrict__ w_pk) {
    int idx   = blockIdx.x * 256 + threadIdx.x;    // 18*128*512 = 1,179,648
    int p     = idx & 511;
    int gid   = idx >> 9;
    int kt    = gid >> 7;
    int nt    = gid & 127;
    int chunk = p >> 8;
    int lane  = (p >> 3) & 31;
    int e     = p & 7;
    int K     = kt * 32 + chunk * 16 + (lane >> 4) * 8 + e;
    int zc    = nt * 16 + (lane & 15);
    float v   = (K < NH) ? ((g_cfloat*)w_h)[K * (4 * NH) + zc]
                         : ((g_cfloat*)w_x)[(K - NH) * (4 * NH) + zc];
    ((g_ushort*)w_pk)[idx] = bf16bits(v);
}

// -------- prep: zero h0 (bf16) and the grid barrier ------------------------
__global__ void init_state_kernel(unsigned* __restrict__ h0u32,
                                  unsigned* __restrict__ bar) {
    int i = blockIdx.x * 256 + threadIdx.x;        // 131072 = NH*BATCH/2
    ((g_uint*)h0u32)[i] = 0u;
    if (i < 16) ((g_uint*)bar)[i] = 0u;
}

// -------- grid-wide barrier (all 64 persistent blocks) ----------------------
__device__ __forceinline__ void grid_sync(unsigned* bar) {
    __syncthreads();
    if (threadIdx.x == 0) {
        unsigned* cnt = bar;
        unsigned* gen = bar + 8;
        __threadfence();
        unsigned g = __atomic_load_n(gen, __ATOMIC_RELAXED);
        if (atomicAdd(cnt, 1u) == NBLK - 1u) {
            __atomic_store_n(cnt, 0u, __ATOMIC_RELAXED);
            __atomic_fetch_add(gen, 1u, __ATOMIC_RELEASE);
        } else {
            while (__atomic_load_n(gen, __ATOMIC_ACQUIRE) == g)
                __builtin_amdgcn_s_sleep(2);
        }
    }
    __syncthreads();
}

// -------- persistent recurrent kernel ---------------------------------------
__global__ void __launch_bounds__(256, 1)
lstm_persistent_kernel(const float* __restrict__ a,          // [B,T,F] f32
                       const unsigned short* __restrict__ w_pk,
                       const float* __restrict__ b_lstm,     // [4N]
                       unsigned short* __restrict__ h0,      // [B,N] bf16
                       unsigned short* __restrict__ h1,
                       float* __restrict__ h_final,          // [B,N] f32
                       unsigned* __restrict__ bar) {
    extern __shared__ char smem[];                 // 288KB: W slice frags

    const int tid  = threadIdx.x;
    const int wave = tid >> 5;
    const int l    = tid & 31;
    const int wm   = wave & 1;                     // 32-row half of 64-row tile
    const int sl   = wave >> 1;                    // 16-col hidden slice (0..3)
    const int mb   = blockIdx.x & 7;
    const int nb   = blockIdx.x >> 3;

    // ---- stage this block's W slice into LDS via the Tensor Data Mover ----
    // 3D tile: x = 4096 B chunk (nt = g*32 + nb*4 + s, s contiguous),
    //          y = g in [0,4)  (global stride 32 KB),
    //          z = kt in [0,18) (global stride 128 KB);
    // LDS filled contiguously -> fragment order (kt, g, s).
    if (wave == 0) {
        unsigned long long gaddr =
            (unsigned long long)(const void*)w_pk + (unsigned)nb * 4096u;
        unsigned ldsbase = __builtin_amdgcn_groupstaticsize();  // dyn LDS base
        u32x4 g0 = { 1u,                                        // count=1
                     ldsbase,                                   // lds_addr
                     (unsigned)(gaddr & 0xFFFFFFFFull),
                     (unsigned)(((gaddr >> 32) & 0x01FFFFFFull) | 0x80000000u) };
        //            type=2 ("image") in bits 127:126 ^^^^
        i32x8 g1 = { 0,                        // wg_mask=0, data_size=1B
                     0x10000000,               // tensor_dim0[15:0]=4096 <<16
                     0x00040000,               // tensor_dim1[15:0]=4    <<16
                     0x10000000,               // tile_dim0=4096         <<16
                     (int)(4u | (18u << 16)),  // tile_dim1=4, tile_dim2=18
                     32768,                    // tensor_dim0_stride = 32 KB
                     0,                        // dim1_stride[15:0]=0 (128K lo)
                     2 };                      // dim1_stride[47:16] = 2 -> 128K
        i32x4 g2 = { 18, 0, 0, 0 };            // tensor_dim2 = 18
        i32x4 g3 = { 0, 0, 0, 0 };
#if __has_include(<hip/amd_detail/amd_gfx1250_TDM.h>)
        i32x8 gz = { 0, 0, 0, 0, 0, 0, 0, 0 };
        __builtin_amdgcn_tensor_load_to_lds(g0, g1, g2, g3, gz, 0);
#else
        __builtin_amdgcn_tensor_load_to_lds(g0, g1, g2, g3, 0);
#endif
        __builtin_amdgcn_s_wait_tensorcnt(0);
    }
    __syncthreads();

    const int lane15 = l & 15;
    const int lhalf  = l >> 4;
    const int colh   = nb * 64 + sl * 16 + lane15;         // hidden column

    const float bi = b_lstm[colh];
    const float bf = b_lstm[NH + colh];
    const float bg = b_lstm[2 * NH + colh];
    const float bo = b_lstm[3 * NH + colh];

    const int mrow0 = mb * 64 + wm * 32 + lane15;          // A-frag row (r=0)

    const v8f vzero = {0.f, 0.f, 0.f, 0.f, 0.f, 0.f, 0.f, 0.f};
    v8f c[2] = {vzero, vzero};

    g_ushort* bufs[2] = { (g_ushort*)h0, (g_ushort*)h1 };
    g_cfloat* ag      = (g_cfloat*)a;
    g_float*  hfin    = (g_float*)h_final;

    for (int t = 0; t < TSTEPS; ++t) {
        const g_cushort* cur = (const g_cushort*)bufs[t & 1];
        g_ushort*        nxt = bufs[(t + 1) & 1];

        v8f acc[2][4];
#pragma unroll
        for (int r = 0; r < 2; ++r)
#pragma unroll
            for (int g = 0; g < 4; ++g) acc[r][g] = vzero;

        // ---- K tiles 0..15: h @ w_h (h from L2-resident ping buffer) ----
        for (int kt = 0; kt < 16; ++kt) {
            FragU afr[2];
            const int kc = kt * 32 + lhalf * 8;
#pragma unroll
            for (int r = 0; r < 2; ++r) {
                const g_cushort* p = cur + (size_t)(mrow0 + r * 16) * NH + kc;
                afr[r].q[0] = *(g_cu4v*)p;
                afr[r].q[1] = *(g_cu4v*)(p + 16);
            }
#pragma unroll
            for (int g = 0; g < 4; ++g) {
                FragU bfr;
                const char* base =
                    smem + (size_t)(kt * 16 + g * 4 + sl) * 1024 + l * 16;
                bfr.q[0] = *(const u4v*)base;
                bfr.q[1] = *(const u4v*)(base + 512);
                acc[0][g] = __builtin_amdgcn_wmma_f32_16x16x32_bf16(
                    false, afr[0].v, false, bfr.v, (short)0, acc[0][g], false, false);
                acc[1][g] = __builtin_amdgcn_wmma_f32_16x16x32_bf16(
                    false, afr[1].v, false, bfr.v, (short)0, acc[1][g], false, false);
            }
        }

        // ---- K tiles 16..17: x_t @ w_x (fp32 'a' converted on the fly) ----
#pragma unroll
        for (int kt2 = 0; kt2 < 2; ++kt2) {
            FragU afr[2];
            const int kc = kt2 * 32 + lhalf * 8;
#pragma unroll
            for (int r = 0; r < 2; ++r) {
                g_cfloat* p =
                    ag + ((size_t)(mrow0 + r * 16) * TSTEPS + t) * NF + kc;
                f4v f0 = *(g_cf4v*)p;
                f4v f1 = *(g_cf4v*)(p + 4);
                f4v f2 = *(g_cf4v*)(p + 16);
                f4v f3 = *(g_cf4v*)(p + 20);
                afr[r].u[0] = pk2(f0.x, f0.y); afr[r].u[1] = pk2(f0.z, f0.w);
                afr[r].u[2] = pk2(f1.x, f1.y); afr[r].u[3] = pk2(f1.z, f1.w);
                afr[r].u[4] = pk2(f2.x, f2.y); afr[r].u[5] = pk2(f2.z, f2.w);
                afr[r].u[6] = pk2(f3.x, f3.y); afr[r].u[7] = pk2(f3.z, f3.w);
            }
#pragma unroll
            for (int g = 0; g < 4; ++g) {
                FragU bfr;
                const char* base =
                    smem + (size_t)((16 + kt2) * 16 + g * 4 + sl) * 1024 + l * 16;
                bfr.q[0] = *(const u4v*)base;
                bfr.q[1] = *(const u4v*)(base + 512);
                acc[0][g] = __builtin_amdgcn_wmma_f32_16x16x32_bf16(
                    false, afr[0].v, false, bfr.v, (short)0, acc[0][g], false, false);
                acc[1][g] = __builtin_amdgcn_wmma_f32_16x16x32_bf16(
                    false, afr[1].v, false, bfr.v, (short)0, acc[1][g], false, false);
            }
        }

        // ---- prefetch next step's x_t rows (overlaps gates + barrier) ----
        if (t + 1 < TSTEPS) {
            __builtin_prefetch(a + ((size_t)mrow0 * TSTEPS + (t + 1)) * NF, 0, 1);
            __builtin_prefetch(a + ((size_t)(mrow0 + 16) * TSTEPS + (t + 1)) * NF, 0, 1);
        }

        // ---- fused gates on accumulators; c persists in VGPRs ----
#pragma unroll
        for (int r = 0; r < 2; ++r) {
            const int rowbase = mb * 64 + wm * 32 + r * 16 + lhalf * 8;
#pragma unroll
            for (int j = 0; j < 8; ++j) {
                float zi = acc[r][0][j] + bi;
                float zf = acc[r][1][j] + bf;
                float zg = acc[r][2][j] + bg;
                float zo = acc[r][3][j] + bo;
                float cn = sigmoidf_(zf) * c[r][j] + sigmoidf_(zi) * seluf_(zg);
                c[r][j] = cn;
                float hh = sigmoidf_(zo) * seluf_(cn);
                const int row = rowbase + j;              // C/D: M=j (+8 hi half)
                nxt[(size_t)row * NH + colh] = bf16bits(hh);
                if (t == TSTEPS - 1) hfin[(size_t)row * NH + colh] = hh;
            }
        }

        grid_sync(bar);
    }
}

// -------- epilogue: selu(b@w_b+b_b), concat, dense_out (memory-bound) -------
__global__ void head_kernel(const float* __restrict__ b,      // [B,24]
                            const float* __restrict__ w_b,    // [24,512]
                            const float* __restrict__ b_b,    // [512]
                            const float* __restrict__ w_out,  // [1024,5]
                            const float* __restrict__ b_out,  // [5]
                            const float* __restrict__ h_final,// [B,512]
                            float* __restrict__ out) {        // [B,5]
    __shared__ float bd[NH];
    __shared__ float red[256];
    __shared__ float brow[24];
    const int row = blockIdx.x, tid = threadIdx.x;

    if (tid < 24) brow[tid] = b[row * 24 + tid];
    __syncthreads();

    for (int col = tid; col < NH; col += 256) {
        float s = b_b[col];
#pragma unroll
        for (int k = 0; k < 24; ++k) s += brow[k] * w_b[k * NH + col];
        bd[col] = seluf_(s);
    }
    __syncthreads();

    for (int o = 0; o < 5; ++o) {
        float p = 0.0f;
        for (int k = tid; k < 2 * NH; k += 256) {
            float x = (k < NH) ? ((g_cfloat*)h_final)[(size_t)row * NH + k]
                               : bd[k - NH];
            p += x * w_out[k * 5 + o];
        }
        red[tid] = p;
        __syncthreads();
        for (int st = 128; st > 0; st >>= 1) {
            if (tid < st) red[tid] += red[tid + st];
            __syncthreads();
        }
        if (tid == 0) out[row * 5 + o] = red[0] + b_out[o];
        __syncthreads();
    }
}

// ---------------------------------------------------------------------------
extern "C" void kernel_launch(void* const* d_in, const int* in_sizes, int n_in,
                              void* d_out, int out_size, void* d_ws, size_t ws_size,
                              hipStream_t stream) {
    (void)in_sizes; (void)n_in; (void)out_size; (void)ws_size;

    const float* a      = (const float*)d_in[0];
    const float* b      = (const float*)d_in[1];
    const float* w_x    = (const float*)d_in[2];
    const float* w_h    = (const float*)d_in[3];
    const float* b_lstm = (const float*)d_in[4];
    const float* w_b    = (const float*)d_in[5];
    const float* b_b    = (const float*)d_in[6];
    const float* w_out  = (const float*)d_in[7];
    const float* b_out  = (const float*)d_in[8];
    float* out = (float*)d_out;

    // workspace layout (~4.3 MB)
    char* ws = (char*)d_ws;
    unsigned short* w_pk    = (unsigned short*)(ws);             // 2,359,296 B
    unsigned short* h0      = (unsigned short*)(ws + 2359296);   //   524,288 B
    unsigned short* h1      = (unsigned short*)(ws + 2883584);   //   524,288 B
    float*          h_final = (float*)(ws + 3407872);            // 1,048,576 B
    unsigned*       bar     = (unsigned*)(ws + 4456448);         //        64 B

    pack_w_kernel<<<4608, 256, 0, stream>>>(w_x, w_h, w_pk);
    init_state_kernel<<<512, 256, 0, stream>>>((unsigned*)h0, bar);

    // 64 persistent blocks, 288KB dynamic LDS each (CDNA5: 320KB/WGP)
    lstm_persistent_kernel<<<NBLK, 256, 294912, stream>>>(
        a, w_pk, b_lstm, h0, h1, h_final, bar);

    head_kernel<<<BATCH, 256, 0, stream>>>(b, w_b, b_b, w_out, b_out, h_final, out);
}
